// LinkWIRE_70377334112504
// MI455X (gfx1250) — matrix operand order repaired and checked by reading, not compiled, not run on hardware
//
#include <hip/hip_runtime.h>
#include <hip/hip_bf16.h>

static constexpr int       NN  = 6144;   // nodes
static constexpr int       INC = 128;    // in channels
static constexpr int       HC  = 256;    // hidden
static constexpr int       KCC = 64;     // centers
static constexpr long long EC  = 196608; // edges

typedef __attribute__((ext_vector_type(16))) __bf16 v16bf;
typedef __attribute__((ext_vector_type(8)))  __bf16 v8bf;
typedef __attribute__((ext_vector_type(8)))  float  v8f;

__device__ __forceinline__ float bf2f(unsigned short h) { return __uint_as_float(((unsigned int)h) << 16); }
__device__ __forceinline__ unsigned short f2bf(float f) {
  unsigned int u = __float_as_uint(f);
  u += 0x7FFFu + ((u >> 16) & 1u);     // round-to-nearest-even
  return (unsigned short)(u >> 16);
}

// ---------------------------------------------------------------- utility
__global__ void k_zero_u32(unsigned int* __restrict__ p, long long n) {
  long long i = (long long)blockIdx.x * blockDim.x + threadIdx.x;
  if (i < n) p[i] = 0u;
}

__global__ void k_cvt_bf(unsigned short* __restrict__ dst, const float* __restrict__ src, long long n) {
  long long i = (long long)blockIdx.x * blockDim.x + threadIdx.x;
  if (i < n) dst[i] = f2bf(src[i]);
}

// dst[Cc,R] = bf16(src[R,Cc])^T   (stage B operands transposed)
__global__ void k_cvt_bf_t(unsigned short* __restrict__ dst, const float* __restrict__ src,
                           int R, int Cc) {
  long long idx = (long long)blockIdx.x * blockDim.x + threadIdx.x;
  if (idx >= (long long)R * Cc) return;
  int r = (int)(idx / Cc), c = (int)(idx % Cc);
  dst[(long long)c * R + r] = f2bf(src[idx]);
}

__global__ void k_scatter_edges(unsigned short* __restrict__ A, const long long* __restrict__ ei) {
  long long e = (long long)blockIdx.x * blockDim.x + threadIdx.x;
  if (e >= EC) return;
  long long r = ei[e];
  long long c = ei[EC + e];
  A[r * (long long)NN + c] = 0x3F80; // bf16 1.0
}

// ------------------------------------------------- WMMA bf16 GEMM (C f32)
// C[M,Nn] = A[M,Kd] * B[Kd,Nn].  A row-major [M,Kd] (lda), B given TRANSPOSED
// row-major Bt[Nn,Kd] (ldbt).  One wave computes a 16x64 C strip (4 WMMA
// tiles sharing one A fragment).  Fragments load as contiguous b128 pairs:
//   A lane(mr,hf): elems [8hf..8hf+7] and [16+8hf..16+8hf+7] of row (K dim)
//   B lane(mr,hf): elems [8hf..8hf+7] and [16+8hf..16+8hf+7] of Bt row `col` (K dim)
__global__ void k_gemm_wmma_bf16(const unsigned short* __restrict__ A,
                                 const unsigned short* __restrict__ Bt,
                                 float* __restrict__ C,
                                 int M, int Nn, int Kd,
                                 int lda, int ldbt, int ldc) {
  int wave    = blockIdx.x * (blockDim.x >> 5) + (threadIdx.x >> 5);
  int tiles_m = M >> 4, nblk = Nn >> 6;
  if (wave >= tiles_m * nblk) return;
  int tm = wave % tiles_m, tnb = wave / tiles_m;
  int lane = threadIdx.x & 31;
  int hf   = lane >> 4;
  int mr   = lane & 15;
  int row  = tm * 16 + mr;
  int col0 = tnb * 64 + mr;

  const __bf16* Ap = (const __bf16*)A  + (long long)row  * lda  + 8  * hf;
  const __bf16* Bp = (const __bf16*)Bt + (long long)col0 * ldbt + 8  * hf;

  v8f acc[4];
#pragma unroll
  for (int j = 0; j < 4; ++j) acc[j] = (v8f){0.f, 0.f, 0.f, 0.f, 0.f, 0.f, 0.f, 0.f};

  for (int k0 = 0; k0 < Kd; k0 += 32) {
    __builtin_prefetch((const void*)(Ap + 256), 0, 0);   // stream-ahead on A
    v8bf a0 = *(const v8bf*)(Ap);
    v8bf a1 = *(const v8bf*)(Ap + 16);
    v16bf af = __builtin_shufflevector(a0, a1, 0, 1, 2, 3, 4, 5, 6, 7,
                                               8, 9, 10, 11, 12, 13, 14, 15);
#pragma unroll
    for (int j = 0; j < 4; ++j) {
      const __bf16* bp = Bp + (long long)j * 16 * ldbt;
      v8bf b0 = *(const v8bf*)(bp);
      v8bf b1 = *(const v8bf*)(bp + 16);
      v16bf bfv = __builtin_shufflevector(b0, b1, 0, 1, 2, 3, 4, 5, 6, 7,
                                                  8, 9, 10, 11, 12, 13, 14, 15);
      acc[j] = __builtin_amdgcn_wmma_f32_16x16x32_bf16(false, af, false, bfv,
                                                       (short)0, acc[j], false, false);
    }
    Ap += 32;
    Bp += 32;
  }
#pragma unroll
  for (int j = 0; j < 4; ++j) {
#pragma unroll
    for (int g = 0; g < 8; ++g) {
      int r = tm * 16 + g + 8 * hf;                      // C layout: VGPR g -> M=g+8*hf
      C[(long long)r * ldc + col0 + 16 * j] = acc[j][g];
    }
  }
}

// ----------------------------------------------------------- elementwise
__global__ void k_bias_cvt(const float* __restrict__ src, const float* __restrict__ bias,
                           unsigned short* __restrict__ dst, long long n, int cols) {
  long long i = (long long)blockIdx.x * blockDim.x + threadIdx.x;
  if (i < n) dst[i] = f2bf(src[i] + bias[i % cols]);
}

// softmax over K=64 per row; writes s (f32), s_bf [N,K], s_bfT [K,N],
// snorm=||s_i||^2, and column sums Ssum (atomics)
__global__ void k_softmax64(const float* __restrict__ slog, const float* __restrict__ bias,
                            float* __restrict__ s, unsigned short* __restrict__ s_bf,
                            unsigned short* __restrict__ s_bfT,
                            float* __restrict__ snorm, float* __restrict__ Ssum) {
  int i = blockIdx.x, k = threadIdx.x;   // blockDim 64
  __shared__ float red[64];
  float v = slog[(long long)i * KCC + k] + bias[k];
  red[k] = v; __syncthreads();
  for (int off = 32; off > 0; off >>= 1) { if (k < off) red[k] = fmaxf(red[k], red[k + off]); __syncthreads(); }
  float mx = red[0]; __syncthreads();
  float e = __expf(v - mx);
  red[k] = e; __syncthreads();
  for (int off = 32; off > 0; off >>= 1) { if (k < off) red[k] += red[k + off]; __syncthreads(); }
  float sm = red[0]; __syncthreads();
  float sv = e / sm;
  unsigned short sb = f2bf(sv);
  s[(long long)i * KCC + k]     = sv;
  s_bf[(long long)i * KCC + k]  = sb;
  s_bfT[(long long)k * NN + i]  = sb;
  red[k] = sv * sv; __syncthreads();
  for (int off = 32; off > 0; off >>= 1) { if (k < off) red[k] += red[k + off]; __syncthreads(); }
  if (k == 0) snorm[i] = red[0];
  atomicAdd(&Ssum[k], sv);
}

__global__ void k_rowsum_diag(const unsigned short* __restrict__ A,
                              float* __restrict__ deg, float* __restrict__ diagA) {
  int i = blockIdx.x;
  __shared__ float red[256];
  float sum = 0.f;
  for (int j = threadIdx.x; j < NN; j += 256) sum += bf2f(A[(long long)i * NN + j]);
  red[threadIdx.x] = sum; __syncthreads();
  for (int off = 128; off > 0; off >>= 1) { if (threadIdx.x < off) red[threadIdx.x] += red[threadIdx.x + off]; __syncthreads(); }
  if (threadIdx.x == 0) { deg[i] = red[0]; diagA[i] = bf2f(A[(long long)i * NN + i]); }
}

// scal[0] += sum_k s[i,k]*t[i,k] (=tr(s^T A s)); scal[1] += deg[i]*||s_i||^2
__global__ void k_traces(const float* __restrict__ s, const float* __restrict__ t,
                         const float* __restrict__ deg, const float* __restrict__ snorm,
                         float* __restrict__ scal) {
  int i = blockIdx.x, k = threadIdx.x; // blockDim 64
  __shared__ float red[64];
  red[k] = s[(long long)i * KCC + k] * t[(long long)i * KCC + k]; __syncthreads();
  for (int off = 32; off > 0; off >>= 1) { if (k < off) red[k] += red[k + off]; __syncthreads(); }
  if (k == 0) { atomicAdd(&scal[0], red[0]); atomicAdd(&scal[1], deg[i] * snorm[i]); }
}

__global__ void k_ss(const float* __restrict__ s, float* __restrict__ ss) {
  int ab = blockIdx.x, a = ab >> 6, b = ab & 63;
  __shared__ float red[256];
  float sum = 0.f;
  for (int i = threadIdx.x; i < NN; i += 256)
    sum += s[(long long)i * KCC + a] * s[(long long)i * KCC + b];
  red[threadIdx.x] = sum; __syncthreads();
  for (int off = 128; off > 0; off >>= 1) { if (threadIdx.x < off) red[threadIdx.x] += red[threadIdx.x + off]; __syncthreads(); }
  if (threadIdx.x == 0) ss[ab] = red[0];
}

__global__ void k_losses(const float* __restrict__ ss, const float* __restrict__ scal,
                         float* __restrict__ out_loss) {
  __shared__ float red[256];
  __shared__ float fro;
  int tid = threadIdx.x;
  float sum = 0.f;
  for (int i = tid; i < KCC * KCC; i += 256) { float v = ss[i]; sum += v * v; }
  red[tid] = sum; __syncthreads();
  for (int off = 128; off > 0; off >>= 1) { if (tid < off) red[tid] += red[tid + off]; __syncthreads(); }
  if (tid == 0) fro = sqrtf(red[0]);
  __syncthreads();
  float f = fro;
  sum = 0.f;
  for (int i = tid; i < KCC * KCC; i += 256) {
    int a = i >> 6, b = i & 63;
    float v = ss[i] / f - ((a == b) ? 0.125f : 0.f);  // 1/sqrt(64)
    sum += v * v;
  }
  red[tid] = sum; __syncthreads();
  for (int off = 128; off > 0; off >>= 1) { if (tid < off) red[tid] += red[tid + off]; __syncthreads(); }
  if (tid == 0) out_loss[0] = -scal[0] / scal[1] + sqrtf(red[0]);
}

// d for GCN1 (A with diag forced to 1): d = clip(deg - diagA + 1, 1)^-1/2
__global__ void k_prep_dg(const float* __restrict__ deg, const float* __restrict__ diagA,
                          float* __restrict__ dg) {
  int i = blockIdx.x * blockDim.x + threadIdx.x;
  if (i < NN) dg[i] = rsqrtf(fmaxf(deg[i] - diagA[i] + 1.f, 1.f));
}

// dst[H,N] (transposed, as B operand) = bf16(dg[i]*src[i,h])
__global__ void k_scale_rows_bf_t(const float* __restrict__ src, const float* __restrict__ dg,
                                  unsigned short* __restrict__ dst_t) {
  long long idx = (long long)blockIdx.x * blockDim.x + threadIdx.x;
  if (idx >= (long long)NN * HC) return;
  long long i = idx / HC; int h = (int)(idx % HC);
  dst_t[(long long)h * NN + i] = f2bf(dg[i] * src[idx]);
}

// z = relu(d*(t1 + (1-diagA)*d*xw) + b1), stored bf16 row-major (A operand)
__global__ void k_gcn1_post(const float* __restrict__ t1, const float* __restrict__ xw,
                            const float* __restrict__ dg, const float* __restrict__ diagA,
                            const float* __restrict__ b1, unsigned short* __restrict__ z_bf,
                            long long n, int cols) {
  long long idx = (long long)blockIdx.x * blockDim.x + threadIdx.x;
  if (idx >= n) return;
  long long i = idx / cols; int h = (int)(idx % cols);
  float tt = t1[idx] + (1.f - diagA[i]) * dg[i] * xw[idx];
  float zz = fmaxf(dg[i] * tt + b1[h], 0.f);
  z_bf[idx] = f2bf(zz);
}

// d2 = clip(s_i . Ssum - ||s_i||^2 + 1, 1)^-1/2  (rowsum of dist with unit diag)
__global__ void k_prep_d2(const float* __restrict__ s, const float* __restrict__ Ssum,
                          const float* __restrict__ snorm, float* __restrict__ d2) {
  int i = blockIdx.x * blockDim.x + threadIdx.x;
  if (i >= NN) return;
  float r = 0.f;
#pragma unroll 8
  for (int k = 0; k < KCC; ++k) r += s[(long long)i * KCC + k] * Ssum[k];
  d2[i] = rsqrtf(fmaxf(r - snorm[i] + 1.f, 1.f));
}

// out = d2*(v + (1-||s_i||^2)*d2*Y) + b2
__global__ void k_gcn2_post(const float* __restrict__ v, const float* __restrict__ Y,
                            const float* __restrict__ d2, const float* __restrict__ snorm,
                            const float* __restrict__ b2, float* __restrict__ out,
                            long long n, int cols) {
  long long idx = (long long)blockIdx.x * blockDim.x + threadIdx.x;
  if (idx >= n) return;
  long long i = idx / cols; int h = (int)(idx % cols);
  out[idx] = d2[i] * (v[idx] + (1.f - snorm[i]) * d2[i] * Y[idx]) + b2[h];
}

// ---------------------------------------------------------------- launch
extern "C" void kernel_launch(void* const* d_in, const int* in_sizes, int n_in,
                              void* d_out, int out_size, void* d_ws, size_t ws_size,
                              hipStream_t stream) {
  const float*     x     = (const float*)d_in[0];
  const long long* ei    = (const long long*)d_in[1];
  const float*     Wmlp  = (const float*)d_in[3];
  const float*     bmlp  = (const float*)d_in[4];
  const float*     Wmlp2 = (const float*)d_in[5];
  const float*     bmlp2 = (const float*)d_in[6];
  const float*     W1    = (const float*)d_in[7];
  const float*     b1    = (const float*)d_in[8];
  const float*     W2    = (const float*)d_in[9];
  const float*     b2    = (const float*)d_in[10];
  float* out = (float*)d_out;

  char* wsp = (char*)d_ws;
  size_t off = 0;
  auto alloc = [&](size_t bytes) -> void* {
    void* p = wsp + off;
    off += (bytes + 255) & ~(size_t)255;
    return p;
  };
  unsigned short* A_bf   = (unsigned short*)alloc((size_t)NN * NN * 2);
  unsigned short* x_bf   = (unsigned short*)alloc((size_t)NN * INC * 2);
  unsigned short* WmlpT  = (unsigned short*)alloc((size_t)NN * HC * 2);   // [HC,NN]
  unsigned short* Wm2T   = (unsigned short*)alloc((size_t)HC * KCC * 2);  // [KCC,HC]
  unsigned short* W1T    = (unsigned short*)alloc((size_t)INC * HC * 2);  // [HC,INC]
  unsigned short* W2T    = (unsigned short*)alloc((size_t)HC * HC * 2);   // [HC,HC]
  float*          H1     = (float*)alloc((size_t)NN * HC * 4);
  unsigned short* H1_bf  = (unsigned short*)alloc((size_t)NN * HC * 2);
  float*          slog   = (float*)alloc((size_t)NN * KCC * 4);
  float*          s      = (float*)alloc((size_t)NN * KCC * 4);
  unsigned short* s_bf   = (unsigned short*)alloc((size_t)NN * KCC * 2);  // [NN,KCC]
  unsigned short* s_bfT  = (unsigned short*)alloc((size_t)NN * KCC * 2);  // [KCC,NN]
  float*          snorm  = (float*)alloc((size_t)NN * 4);
  float*          Ssum   = (float*)alloc((size_t)KCC * 4);
  float*          t      = (float*)alloc((size_t)NN * KCC * 4);
  float*          deg    = (float*)alloc((size_t)NN * 4);
  float*          diagA  = (float*)alloc((size_t)NN * 4);
  float*          dg     = (float*)alloc((size_t)NN * 4);
  float*          xw     = (float*)alloc((size_t)NN * HC * 4);
  unsigned short* yT_bf  = (unsigned short*)alloc((size_t)NN * HC * 2);   // [HC,NN]
  float*          t1     = (float*)alloc((size_t)NN * HC * 4);
  unsigned short* z_bf   = (unsigned short*)alloc((size_t)NN * HC * 2);
  float*          Y      = (float*)alloc((size_t)NN * HC * 4);
  float*          d2     = (float*)alloc((size_t)NN * 4);
  unsigned short* y2T_bf = (unsigned short*)alloc((size_t)NN * HC * 2);   // [HC,NN]
  float*          u      = (float*)alloc((size_t)KCC * HC * 4);
  unsigned short* uT_bf  = (unsigned short*)alloc((size_t)KCC * HC * 2);  // [HC,KCC]
  float*          v      = (float*)alloc((size_t)NN * HC * 4);
  float*          ss     = (float*)alloc((size_t)KCC * KCC * 4);
  float*          scal   = (float*)alloc((size_t)16 * 4);

  // gemm: C[M,Nn] = A[M,Kd] @ B, with B passed transposed Bt[Nn,Kd]
  auto gemm = [&](const unsigned short* Ag, const unsigned short* Btg, float* Cg,
                  int M, int Nn, int Kd, int lda, int ldbt, int ldc) {
    int waves  = (M / 16) * (Nn / 64);
    int blocks = (waves + 7) / 8;                 // 8 waves (256 thr) per block
    k_gemm_wmma_bf16<<<blocks, 256, 0, stream>>>(Ag, Btg, Cg, M, Nn, Kd, lda, ldbt, ldc);
  };
  const long long NH = (long long)NN * HC;

  // -- build A (bf16, exact 0/1) ------------------------------------------
  long long a_u32 = (long long)NN * NN / 2;
  k_zero_u32<<<(int)((a_u32 + 255) / 256), 256, 0, stream>>>((unsigned int*)A_bf, a_u32);
  k_zero_u32<<<1, 256, 0, stream>>>((unsigned int*)Ssum, KCC);
  k_zero_u32<<<1, 32, 0, stream>>>((unsigned int*)scal, 16);
  k_scatter_edges<<<(int)((EC + 255) / 256), 256, 0, stream>>>(A_bf, ei);

  // -- bf16 operand staging (B operands transposed) -----------------------
  k_cvt_bf_t<<<(int)((NH + 255) / 256), 256, 0, stream>>>(WmlpT, Wmlp, NN, HC);
  k_cvt_bf_t<<<(HC * KCC + 255) / 256, 256, 0, stream>>>(Wm2T, Wmlp2, HC, KCC);
  k_cvt_bf_t<<<(INC * HC + 255) / 256, 256, 0, stream>>>(W1T, W1, INC, HC);
  k_cvt_bf_t<<<(HC * HC + 255) / 256, 256, 0, stream>>>(W2T, W2, HC, HC);
  k_cvt_bf<<<(int)(((long long)NN * INC + 255) / 256), 256, 0, stream>>>(x_bf, x, (long long)NN * INC);

  // -- cluster logits + softmax -------------------------------------------
  gemm(A_bf, WmlpT, H1, NN, HC, NN, NN, NN, HC);                      // A @ W_mlp
  k_bias_cvt<<<(int)((NH + 255) / 256), 256, 0, stream>>>(H1, bmlp, H1_bf, NH, HC);
  gemm(H1_bf, Wm2T, slog, NN, KCC, HC, HC, HC, KCC);                  // (.+b) @ W_mlp2
  k_softmax64<<<NN, 64, 0, stream>>>(slog, bmlp2, s, s_bf, s_bfT, snorm, Ssum);

  // -- pump losses ---------------------------------------------------------
  k_rowsum_diag<<<NN, 256, 0, stream>>>(A_bf, deg, diagA);
  gemm(A_bf, s_bfT, t, NN, KCC, NN, NN, NN, KCC);                     // A @ s
  k_traces<<<NN, 64, 0, stream>>>(s, t, deg, snorm, scal);
  k_ss<<<KCC * KCC, 256, 0, stream>>>(s, ss);
  k_losses<<<1, 256, 0, stream>>>(ss, scal, out + NH);

  // -- GCN layer 1: relu(An @ (x@W1) + b1) --------------------------------
  gemm(x_bf, W1T, xw, NN, HC, INC, INC, INC, HC);                     // x @ W1
  k_prep_dg<<<(NN + 255) / 256, 256, 0, stream>>>(deg, diagA, dg);
  k_scale_rows_bf_t<<<(int)((NH + 255) / 256), 256, 0, stream>>>(xw, dg, yT_bf);
  gemm(A_bf, yT_bf, t1, NN, HC, NN, NN, NN, HC);                      // A @ (d∘xW1)
  k_gcn1_post<<<(int)((NH + 255) / 256), 256, 0, stream>>>(t1, xw, dg, diagA, b1, z_bf, NH, HC);

  // -- GCN layer 2 with rank-64 dist = s s^T (diag->1) ---------------------
  gemm(z_bf, W2T, Y, NN, HC, HC, HC, HC, HC);                         // z @ W2
  k_prep_d2<<<(NN + 255) / 256, 256, 0, stream>>>(s, Ssum, snorm, d2);
  k_scale_rows_bf_t<<<(int)((NH + 255) / 256), 256, 0, stream>>>(Y, d2, y2T_bf);
  gemm(s_bfT, y2T_bf, u, KCC, HC, NN, NN, NN, HC);                    // s^T @ (d2∘Y)
  k_cvt_bf_t<<<(KCC * HC + 255) / 256, 256, 0, stream>>>(uT_bf, u, KCC, HC);
  gemm(s_bf, uT_bf, v, NN, HC, KCC, KCC, KCC, HC);                    // s @ u
  k_gcn2_post<<<(int)((NH + 255) / 256), 256, 0, stream>>>(v, Y, d2, snorm, b2, out, NH, HC);
  (void)in_sizes; (void)n_in; (void)out_size; (void)ws_size;
}
